// ProtEncoder_70506183131680
// MI455X (gfx1250) — matrix operand order repaired and checked by reading, not compiled, not run on hardware
//
#include <hip/hip_runtime.h>
#include <hip/hip_bf16.h>

#define CCH   128           // feature channels
#define KTOT  256           // 2*C (concat of dst||src features)
#define CIN   32
#define CPAIR (2 * KTOT)    // interleaved column pair: [Wf col | Ws col] = 512 bf16

typedef __attribute__((ext_vector_type(16))) __bf16 bf16x16;
typedef __attribute__((ext_vector_type(8)))  __bf16 bf16x8;
typedef __attribute__((ext_vector_type(8)))  float  f32x8;

union AFrag { bf16x16 v; bf16x8 h[2]; };
union BFrag { bf16x16 v; bf16x8 h[2]; };

// ---------------------------------------------------------------- utilities
__global__ void zero_f32(float* __restrict__ p, int n) {
    int i = blockIdx.x * blockDim.x + threadIdx.x;
    if (i < n) p[i] = 0.0f;
}

// degree (segment count) per destination node
__global__ void degree_kernel(const int* __restrict__ dstI, int E,
                              float* __restrict__ deg) {
    int e = blockIdx.x * blockDim.x + threadIdx.x;
    if (e < E) atomicAdd(deg + dstI[e], 1.0f);
}

// transpose + cast one [KTOT x CCH] f32 weight into the interleaved col-major
// bf16 layout: Wt[c * CPAIR + colOff + k]  (colOff = 0 for Wf, KTOT for Ws)
__global__ void prep_w(const float* __restrict__ W, __bf16* __restrict__ Wt,
                       int colOff) {
    int i = blockIdx.x * blockDim.x + threadIdx.x;     // i = k*CCH + c
    if (i >= KTOT * CCH) return;
    int k = i / CCH, c = i % CCH;
    Wt[(size_t)c * CPAIR + colOff + k] = (__bf16)W[i];
}

// ---------------------------------------------------------------- encoder
__global__ void mlp1(const float* __restrict__ x_prot,
                     const float* __restrict__ We1, const float* __restrict__ be1,
                     float* __restrict__ h, int total) {
    int i = blockIdx.x * blockDim.x + threadIdx.x;
    if (i >= total) return;
    int n = i >> 7, j = i & (CCH - 1);
    const float* xr = x_prot + (size_t)n * CIN;
    float acc = be1[j];
#pragma unroll
    for (int k = 0; k < CIN; ++k) acc = fmaf(xr[k], We1[k * CCH + j], acc);
    h[i] = fmaxf(acc, 0.0f);
}

__global__ void mlp2(const float* __restrict__ h,
                     const float* __restrict__ We2, const float* __restrict__ be2,
                     float* __restrict__ x, __bf16* __restrict__ xb, int total) {
    int i = blockIdx.x * blockDim.x + threadIdx.x;
    if (i >= total) return;
    int n = i >> 7, j = i & (CCH - 1);
    const float* hr = h + (size_t)n * CCH;
    float acc = be2[j];
#pragma unroll 8
    for (int k = 0; k < CCH; ++k) acc = fmaf(hr[k], We2[k * CCH + j], acc);
    x[i]  = acc;
    xb[i] = (__bf16)acc;
}

// ---------------------------------------------------------------- CGConv edge GEMM (WMMA)
// One wave handles 32 edges = two 16-row A sub-tiles that SHARE every B
// fragment. Weights live in LDS with per-column interleaving ([Wf|Ws] = 1 KB
// per output channel) so all B loads use immediate DS offsets off one base.
// Per tile: 8 N-tiles x 8 K-steps x 2 matrices x 2 M-subtiles = 256 WMMAs.
__global__ void __launch_bounds__(128)
cgconv_edges(const __bf16* __restrict__ xb,
             const int* __restrict__ ei, int E,
             const __bf16* __restrict__ Wpack,   // interleaved, CCH*CPAIR bf16
             const float* __restrict__ bf, const float* __restrict__ bs,
             float* __restrict__ agg) {
    extern __shared__ __bf16 lds[];

    { // cooperative stage of the packed weights (128 KB) into LDS
        const uint4* g = (const uint4*)Wpack;
        uint4* l = (uint4*)lds;
        const int n16 = CCH * CPAIR * 2 / 16;         // 8192 x 16B
        for (int i = threadIdx.x; i < n16; i += blockDim.x) l[i] = g[i];
    }
    __syncthreads();

    const int lane  = threadIdx.x & 31;
    const int wave  = threadIdx.x >> 5;
    const int wpb   = blockDim.x >> 5;
    const int nTiles = (E + 31) >> 5;
    const int* srcI = ei;          // ei[0] = src
    const int* dstI = ei + E;      // ei[1] = dst  (z = [x[dst], x[src]], agg by dst)

    const int mA   = lane & 15;            // A-matrix row owned by this lane
    const int subA = (lane >> 4) << 3;     // +0 / +8 K sub-block for A
    const int subB = (lane >> 4) << 3;     // +0 / +8 K sub-block for B
    const int mC0  = (lane >> 4) << 3;     // C rows: mC0..mC0+7
    const int colLane = lane & 15;         // C/B column within N-tile

    for (int tile = blockIdx.x * wpb + wave; tile < nTiles;
         tile += gridDim.x * wpb) {
        const int e0 = tile << 5;

        // --- gather A fragments: two 16-edge sub-tiles x 256 K (bf16) ---
        int eA0 = e0 + mA;
        int eA1 = e0 + 16 + mA;
        int ec0 = (eA0 < E) ? eA0 : (E - 1);
        int ec1 = (eA1 < E) ? eA1 : (E - 1);
        int nd0 = dstI[ec0], ns0 = srcI[ec0];
        int nd1 = dstI[ec1], ns1 = srcI[ec1];
        AFrag a0[8], a1[8];
#pragma unroll
        for (int kk = 0; kk < 8; ++kk) {
            int col0 = ((kk & 3) << 5) + subA;         // within-node K base
            int n0 = (kk < 4) ? nd0 : ns0;             // K<128 -> dst half
            int n1 = (kk < 4) ? nd1 : ns1;
            const __bf16* p0 = xb + (size_t)n0 * CCH + col0;
            const __bf16* p1 = xb + (size_t)n1 * CCH + col0;
            a0[kk].h[0] = *(const bf16x8*)(p0);
            a0[kk].h[1] = *(const bf16x8*)(p0 + 16);
            a1[kk].h[0] = *(const bf16x8*)(p1);
            a1[kk].h[1] = *(const bf16x8*)(p1 + 16);
        }

        // --- per-C-row edge ids + dst for aggregation (both sub-tiles) ---
        int eR0[8], dR0[8], eR1[8], dR1[8];
#pragma unroll
        for (int r = 0; r < 8; ++r) {
            int ea = e0 + mC0 + r;
            int eb = e0 + 16 + mC0 + r;
            eR0[r] = ea; dR0[r] = dstI[(ea < E) ? ea : (E - 1)];
            eR1[r] = eb; dR1[r] = dstI[(eb < E) ? eb : (E - 1)];
        }

        // --- 8 N-tiles of 16 output channels ---
#pragma unroll 1
        for (int nt = 0; nt < 8; ++nt) {
            const int cOut = (nt << 4) + colLane;
            f32x8 accF0, accF1, accS0, accS1;
            const float bfvv = bf[cOut], bsvv = bs[cOut];
#pragma unroll
            for (int r = 0; r < 8; ++r) {
                accF0[r] = bfvv; accF1[r] = bfvv;
                accS0[r] = bsvv; accS1[r] = bsvv;
            }

            // one base per column pair; every F/S fragment is base + imm<1024B
            const __bf16* colBase = lds + (size_t)cOut * CPAIR + subB;
#pragma unroll
            for (int kk = 0; kk < 8; ++kk) {
                const int kB = kk << 5;
                BFrag b;
                b.h[0] = *(const bf16x8*)(colBase + kB);          // Wf
                b.h[1] = *(const bf16x8*)(colBase + kB + 16);
                accF0 = __builtin_amdgcn_wmma_f32_16x16x32_bf16(
                    false, a0[kk].v, false, b.v, (short)0, accF0, false, false);
                accF1 = __builtin_amdgcn_wmma_f32_16x16x32_bf16(
                    false, a1[kk].v, false, b.v, (short)0, accF1, false, false);
                b.h[0] = *(const bf16x8*)(colBase + KTOT + kB);   // Ws
                b.h[1] = *(const bf16x8*)(colBase + KTOT + kB + 16);
                accS0 = __builtin_amdgcn_wmma_f32_16x16x32_bf16(
                    false, a0[kk].v, false, b.v, (short)0, accS0, false, false);
                accS1 = __builtin_amdgcn_wmma_f32_16x16x32_bf16(
                    false, a1[kk].v, false, b.v, (short)0, accS1, false, false);
            }

            // msg = sigmoid(f) * softplus(s); scatter-add to agg[dst]
#pragma unroll
            for (int r = 0; r < 8; ++r) {
                if (eR0[r] < E) {
                    float f = accF0[r], s = accS0[r];
                    float sig = 1.0f / (1.0f + __expf(-f));
                    float sp  = fmaxf(s, 0.0f) + __logf(1.0f + __expf(-fabsf(s)));
                    atomicAdd(agg + (size_t)dR0[r] * CCH + cOut, sig * sp);
                }
                if (eR1[r] < E) {
                    float f = accF1[r], s = accS1[r];
                    float sig = 1.0f / (1.0f + __expf(-f));
                    float sp  = fmaxf(s, 0.0f) + __logf(1.0f + __expf(-fabsf(s)));
                    atomicAdd(agg + (size_t)dR1[r] * CCH + cOut, sig * sp);
                }
            }
        }
    }
}

// ---------------------------------------------------------------- residual / mean / relu
__global__ void cgconv_finish(const float* __restrict__ xin,
                              const float* __restrict__ agg,
                              const float* __restrict__ deg,
                              float* __restrict__ xout,
                              __bf16* __restrict__ xbout, int total) {
    int i = blockIdx.x * blockDim.x + threadIdx.x;
    if (i >= total) return;
    int node = i >> 7;
    float cnt = fmaxf(deg[node], 1.0f);
    float v = fmaxf(xin[i] + agg[i] / cnt, 0.0f);
    xout[i] = v;
    if (xbout) xbout[i] = (__bf16)v;
}

// ---------------------------------------------------------------- host orchestration
extern "C" void kernel_launch(void* const* d_in, const int* in_sizes, int n_in,
                              void* d_out, int out_size, void* d_ws, size_t ws_size,
                              hipStream_t stream) {
    const float* x_prot = (const float*)d_in[0];
    const int*   ei_b   = (const int*)d_in[2];
    const int*   ei_r   = (const int*)d_in[3];
    const float* We1 = (const float*)d_in[4];  const float* be1 = (const float*)d_in[5];
    const float* We2 = (const float*)d_in[6];  const float* be2 = (const float*)d_in[7];
    const float* Wf[4]  = {(const float*)d_in[8],  (const float*)d_in[12],
                           (const float*)d_in[16], (const float*)d_in[20]};
    const float* bfv[4] = {(const float*)d_in[9],  (const float*)d_in[13],
                           (const float*)d_in[17], (const float*)d_in[21]};
    const float* Wsm[4] = {(const float*)d_in[10], (const float*)d_in[14],
                           (const float*)d_in[18], (const float*)d_in[22]};
    const float* bsv[4] = {(const float*)d_in[11], (const float*)d_in[15],
                           (const float*)d_in[19], (const float*)d_in[23]};

    const int N  = in_sizes[0] / CIN;
    const int Eb = in_sizes[2] / 2;
    const int Er = in_sizes[3] / 2;
    const int total = N * CCH;

    // carve workspace
    char* ws = (char*)d_ws; size_t off = 0;
    auto carve = [&](size_t bytes) -> void* {
        void* p = ws + off; off = (off + bytes + 255) & ~(size_t)255; return p;
    };
    float*  x    = (float*)carve((size_t)total * 4);
    float*  h    = (float*)carve((size_t)total * 4);
    __bf16* xb   = (__bf16*)carve((size_t)total * 2);
    float*  agg  = (float*)carve((size_t)total * 4);
    float*  degb = (float*)carve((size_t)N * 4);
    float*  degr = (float*)carve((size_t)N * 4);
    __bf16* wt[4];
    for (int l = 0; l < 4; ++l) wt[l] = (__bf16*)carve((size_t)CCH * CPAIR * 2);
    (void)ws_size; (void)n_in;

    const int B256 = 256;
    const int gTot = (total + B256 - 1) / B256;
    const int gW   = (KTOT * CCH + B256 - 1) / B256;

    // 1) weight prep: transpose+cast into interleaved col-major bf16
    for (int l = 0; l < 4; ++l) {
        prep_w<<<gW, B256, 0, stream>>>(Wf[l],  wt[l], 0);
        prep_w<<<gW, B256, 0, stream>>>(Wsm[l], wt[l], KTOT);
    }

    // 2) encoder MLP
    mlp1<<<gTot, B256, 0, stream>>>(x_prot, We1, be1, h, total);
    mlp2<<<gTot, B256, 0, stream>>>(h, We2, be2, x, xb, total);

    // 3) degrees
    zero_f32<<<(N + B256 - 1) / B256, B256, 0, stream>>>(degb, N);
    zero_f32<<<(N + B256 - 1) / B256, B256, 0, stream>>>(degr, N);
    degree_kernel<<<(Eb + B256 - 1) / B256, B256, 0, stream>>>(ei_b + Eb, Eb, degb);
    degree_kernel<<<(Er + B256 - 1) / B256, B256, 0, stream>>>(ei_r + Er, Er, degr);

    // 4) four CGConv layers
    const int*   eis[4] = {ei_b, ei_b, ei_r, ei_r};
    const int    Es[4]  = {Eb, Eb, Er, Er};
    const float* dgs[4] = {degb, degb, degr, degr};
    const size_t ldsB = (size_t)CCH * CPAIR * 2;      // 128 KB

    for (int l = 0; l < 4; ++l) {
        zero_f32<<<gTot, B256, 0, stream>>>(agg, total);
        int nTiles = (Es[l] + 31) / 32;
        int grid = (nTiles + 3) / 4; if (grid > 1024) grid = 1024; if (grid < 1) grid = 1;
        cgconv_edges<<<grid, 128, ldsB, stream>>>(
            xb, eis[l], Es[l], wt[l], bfv[l], bsv[l], agg);
        float*  xo  = (l == 3) ? (float*)d_out : x;
        __bf16* xbo = (l == 3) ? (__bf16*)nullptr : xb;
        cgconv_finish<<<gTot, B256, 0, stream>>>(x, agg, dgs[l], xo, xbo, total);
    }
    (void)out_size;
}
